// ConvE_80814104641908
// MI455X (gfx1250) — matrix operand order repaired and checked by reading, not compiled, not run on hardware
//
#include <hip/hip_runtime.h>

// ---------------------------------------------------------------------------
// Problem constants (match reference)
// ---------------------------------------------------------------------------
#define BATCH 8192
#define DDIM  512
#define TWOD  1024
#define NREL  2048
#define TOPK  16

typedef __attribute__((ext_vector_type(16))) __bf16 bf16x16;
typedef __attribute__((ext_vector_type(8)))  __bf16 bf16x8;
typedef __attribute__((ext_vector_type(4)))  __bf16 bf16x4;
typedef __attribute__((ext_vector_type(8)))  float  f32x8;
typedef __attribute__((ext_vector_type(4)))  unsigned u32x4;
typedef __attribute__((ext_vector_type(8)))  int      i32x8;
typedef __attribute__((ext_vector_type(4)))  int      i32x4;

static __device__ __forceinline__ bf16x16 join8(bf16x8 lo, bf16x8 hi) {
    return __builtin_shufflevector(lo, hi, 0, 1, 2, 3, 4, 5, 6, 7,
                                           8, 9, 10, 11, 12, 13, 14, 15);
}

// LDS byte offset of a generic pointer known to point into LDS.
static __device__ __forceinline__ unsigned lds_byte_off(const void* p) {
    return (unsigned)(unsigned long long)
        (__attribute__((address_space(3))) const char*)p;
}

// ---------------------------------------------------------------------------
// Tensor Data Mover: DMA a contiguous [rows x rowbytes] panel (rowbytes
// contiguous rows, i.e. fully contiguous block) from global to LDS.
// D# layout per CDNA5 ISA ch.8: data_size = 8B, 2-D tile.
// Caller: issue from ONE wave, then s_wait_tensorcnt 0 before the barrier.
// ---------------------------------------------------------------------------
static __device__ __forceinline__ void tdm_load_panel(const void* gsrc,
                                                      unsigned lds_off,
                                                      unsigned rows,
                                                      unsigned rowbytes) {
    const unsigned long long ga = (unsigned long long)gsrc;
    const unsigned elems = rowbytes >> 3;          // 8-byte elements per row

    u32x4 g0;
    g0.x = 1u;                                     // count=1 (valid), user mode
    g0.y = lds_off;                                // lds_addr
    g0.z = (unsigned)(ga & 0xFFFFFFFFu);           // global_addr[31:0]
    g0.w = (unsigned)((ga >> 32) & 0x01FFFFFFu)    // global_addr[56:32]
         | (2u << 30);                             // type = 2 ("image")

    i32x8 g1;
    g1[0] = (int)(3u << 16);                       // wg_mask=0, data_size=3 (8B)
    g1[1] = (int)((elems & 0xFFFFu) << 16);        // tensor_dim0[15:0]
    g1[2] = (int)(((elems >> 16) & 0xFFFFu)        // tensor_dim0[31:16]
         | ((rows & 0xFFFFu) << 16));              // tensor_dim1[15:0]
    g1[3] = (int)(((rows >> 16) & 0xFFFFu)         // tensor_dim1[31:16]
         | ((elems & 0xFFFFu) << 16));             // tile_dim0
    g1[4] = (int)(rows & 0xFFFFu);                 // tile_dim1 (tile_dim2=0)
    g1[5] = (int)elems;                            // tensor_dim0_stride[31:0]
    g1[6] = 0;                                     // stride0[47:32], stride1[15:0]
    g1[7] = 0;                                     // tensor_dim1_stride[47:16]

    i32x4 gz = {0, 0, 0, 0};                       // groups 2/3 unused (2-D)

#if defined(__clang_major__) && (__clang_major__ >= 23)
    i32x8 gz8 = {0, 0, 0, 0, 0, 0, 0, 0};
    __builtin_amdgcn_tensor_load_to_lds(g0, g1, gz, gz, gz8, 0);
#else
    __builtin_amdgcn_tensor_load_to_lds(g0, g1, gz, gz, 0);
#endif
}

// ---------------------------------------------------------------------------
// fp32 -> bf16 conversion kernels
// ---------------------------------------------------------------------------
__global__ void __launch_bounds__(256)
pack_stacked_bf16(const float* __restrict__ e1, const float* __restrict__ e2,
                  __bf16* __restrict__ out /* [BATCH][TWOD] */) {
    int i = (blockIdx.x * 256 + threadIdx.x) * 4;   // element index, multiple of 4
    int row = i >> 10;
    int col = i & 1023;
    const float* src = (col < DDIM) ? (e1 + (size_t)row * DDIM + col)
                                    : (e2 + (size_t)row * DDIM + (col - DDIM));
    float4 v = *(const float4*)src;
    bf16x4 o;
    o.x = (__bf16)v.x; o.y = (__bf16)v.y; o.z = (__bf16)v.z; o.w = (__bf16)v.w;
    *(bf16x4*)(out + i) = o;
}

__global__ void __launch_bounds__(256)
cvt_bf16x4(const float* __restrict__ in, __bf16* __restrict__ out, int n4) {
    int i = blockIdx.x * 256 + threadIdx.x;
    if (i < n4) {
        float4 v = ((const float4*)in)[i];
        bf16x4 o;
        o.x = (__bf16)v.x; o.y = (__bf16)v.y; o.z = (__bf16)v.z; o.w = (__bf16)v.w;
        ((bf16x4*)out)[i] = o;
    }
}

// ---------------------------------------------------------------------------
// R1 = relu(R @ W^T + b)   [NREL x DDIM], bf16 WMMA, fp32 accumulate
// One wave per 16x16 tile; 8 waves/block; 4096 tiles -> 512 blocks.
// ---------------------------------------------------------------------------
__global__ void __launch_bounds__(256)
r1_gemm_kernel(const __bf16* __restrict__ Rb,   // [NREL][TWOD]
               const __bf16* __restrict__ Wb,   // [DDIM][TWOD]
               const float*  __restrict__ bias, // [DDIM]
               float* __restrict__ R1) {        // [NREL][DDIM]
    const int lane = threadIdx.x & 31;
    const int wid  = blockIdx.x * 8 + (threadIdx.x >> 5);  // 0..4095
    const int mt = wid >> 5;     // 0..127
    const int nt = wid & 31;     // 0..31
    const int m0 = mt * 16, n0 = nt * 16;
    const int h = lane >> 4, l15 = lane & 15;

    const __bf16* arow = Rb + (size_t)(m0 + l15) * TWOD;
    const __bf16* brow = Wb + (size_t)(n0 + l15) * TWOD;

    f32x8 acc = {};
    for (int k = 0; k < TWOD; k += 32) {
        // A fragment: 16x32 bf16, ISA layout (lanes<16: K {0..7,16..23}; lanes>=16: +8)
        bf16x8 a_lo = *(const bf16x8*)(arow + k + h * 8);
        bf16x8 a_hi = *(const bf16x8*)(arow + k + 16 + h * 8);
        // B fragment: 32x16 bf16, column-major source rows (lane n, K {0..7,16..23}; lanes>=16: +8)
        bf16x8 b_lo = *(const bf16x8*)(brow + k + h * 8);
        bf16x8 b_hi = *(const bf16x8*)(brow + k + 16 + h * 8);
        acc = __builtin_amdgcn_wmma_f32_16x16x32_bf16(
            false, join8(a_lo, a_hi), false, join8(b_lo, b_hi),
            (short)0, acc, false, false);
    }
    // Epilogue: C/D layout VGPR r -> M = r + 8h, N = l15
#pragma unroll
    for (int r = 0; r < 8; ++r) {
        int m = m0 + r + h * 8;
        int n = n0 + l15;
        float v = acc[r] + bias[n];
        R1[(size_t)m * DDIM + n] = v > 0.0f ? v : 0.0f;
    }
}

// ---------------------------------------------------------------------------
// Fused gating kernel: per block of 16 batch rows:
//   1) TDM-stage A-panel (16x1024 bf16 = 32 KB) into LDS
//   2) WMMA sweep over all 128 N-tiles (4 tiles per A-fragment load),
//      sigmoid -> S_lds (16x2048 f32, LDS only; never touches HBM)
//   3) per-row iterative top-16 (wave argmax + mask)
//   4) out[row] = sum_k Sx_k * R1[idx_k] / sum_k Sx_k
// LDS: 32768 + 131072 + 1024 + 1024 + 64 = 165952 B  (fits in 320 KB WGP LDS)
// ---------------------------------------------------------------------------
#define GATE_LDS_BYTES (32768 + 131072 + 1024 + 1024 + 64)

__global__ void __launch_bounds__(256)
gate_fused_kernel(const __bf16* __restrict__ Xb,  // stacked bf16 [BATCH][TWOD]
                  const __bf16* __restrict__ Rb,  // [NREL][TWOD]
                  const float*  __restrict__ R1,  // [NREL][DDIM]
                  float* __restrict__ out) {      // [BATCH][DDIM]
    extern __shared__ char smem_raw[];
    __bf16* A_lds = (__bf16*)smem_raw;                         // 16*1024 bf16
    float*  S_lds = (float*)(smem_raw + 32768);                // 16*2048 f32
    float*  Sx    = (float*)(smem_raw + 32768 + 131072);       // 16*16
    int*    Ix    = (int*)  (smem_raw + 32768 + 131072 + 1024);// 16*16
    float*  Dn    = (float*)(smem_raw + 32768 + 131072 + 2048);// 16

    const int tid  = threadIdx.x;
    const int lane = tid & 31;
    const int w    = tid >> 5;            // wave 0..7
    const int row0 = blockIdx.x * 16;
    const int h = lane >> 4, l15 = lane & 15;

    // ---- phase 1: Tensor Data Mover: global -> LDS A panel (one wave issues) ----
    if (w == 0) {
        tdm_load_panel(Xb + (size_t)row0 * TWOD, lds_byte_off(A_lds),
                       /*rows=*/16, /*rowbytes=*/TWOD * 2);
        __builtin_amdgcn_s_wait_tensorcnt(0);
    }
    __syncthreads();

    // ---- phase 2: WMMA sweep, 4 N-tiles per A-fragment load ----
    {
        const __bf16* arow = A_lds + (size_t)l15 * TWOD;
        for (int g = 0; g < 4; ++g) {                // 16 tiles/wave in 4 groups
            const int n0base = (w * 16 + g * 4) * 16;
            f32x8 acc[4] = {{}, {}, {}, {}};
            for (int k = 0; k < TWOD; k += 32) {
                bf16x16 af = join8(*(const bf16x8*)(arow + k + h * 8),
                                   *(const bf16x8*)(arow + k + 16 + h * 8));
#pragma unroll
                for (int t = 0; t < 4; ++t) {
                    const __bf16* brow = Rb + (size_t)(n0base + t * 16 + l15) * TWOD;
                    if (t == 0 && k + 256 < TWOD)
                        __builtin_prefetch(brow + k + 256, 0, 1);  // global_prefetch_b8
                    bf16x16 bfm = join8(*(const bf16x8*)(brow + k + h * 8),
                                        *(const bf16x8*)(brow + k + 16 + h * 8));
                    acc[t] = __builtin_amdgcn_wmma_f32_16x16x32_bf16(
                        false, af, false, bfm, (short)0, acc[t], false, false);
                }
            }
#pragma unroll
            for (int t = 0; t < 4; ++t) {
                const int n0 = n0base + t * 16;
#pragma unroll
                for (int r = 0; r < 8; ++r) {
                    int m = r + h * 8;
                    float s = 1.0f / (1.0f + __expf(-acc[t][r]));
                    S_lds[m * NREL + n0 + l15] = s;
                }
            }
        }
    }
    __syncthreads();

    // ---- phase 3: top-16 per row (2 rows per wave) ----
    for (int rr = 0; rr < 2; ++rr) {
        const int m = w * 2 + rr;
        float* Srow = S_lds + m * NREL;
        float dsum = 0.0f;
        for (int sel = 0; sel < TOPK; ++sel) {
            float best = -1.0f;
            int   bi   = 0;
            for (int j = lane; j < NREL; j += 32) {
                float v = Srow[j];
                if (v > best) { best = v; bi = j; }
            }
#pragma unroll
            for (int off = 16; off > 0; off >>= 1) {
                float ov = __shfl_xor(best, off, 32);
                int   oi = __shfl_xor(bi, off, 32);
                if (ov > best || (ov == best && oi < bi)) { best = ov; bi = oi; }
            }
            // best/bi now uniform across the wave
            if (lane == 0) { Sx[m * TOPK + sel] = best; Ix[m * TOPK + sel] = bi; }
            Srow[bi] = -1.0f;   // uniform masking write (sigmoid > 0 always)
            dsum += best;
        }
        if (lane == 0) Dn[m] = dsum;
    }
    __syncthreads();

    // ---- phase 4: gather + weighted sum (R1 is L2-resident, 4 MB) ----
    for (int m = 0; m < 16; ++m) {
        const float inv = 1.0f / Dn[m];
#pragma unroll
        for (int ci = 0; ci < 2; ++ci) {
            const int c = tid + ci * 256;
            float a = 0.0f;
#pragma unroll
            for (int kk = 0; kk < TOPK; ++kk) {
                float wgt = Sx[m * TOPK + kk];
                int   rid = Ix[m * TOPK + kk];
                a += wgt * R1[(size_t)rid * DDIM + c];
            }
            out[(size_t)(row0 + m) * DDIM + c] = a * inv;
        }
    }
}

// ---------------------------------------------------------------------------
// Host launcher
// ---------------------------------------------------------------------------
extern "C" void kernel_launch(void* const* d_in, const int* in_sizes, int n_in,
                              void* d_out, int out_size, void* d_ws, size_t ws_size,
                              hipStream_t stream) {
    (void)in_sizes; (void)n_in; (void)out_size; (void)ws_size;

    const float* e1   = (const float*)d_in[0];  // [B, D]
    const float* e2   = (const float*)d_in[1];  // [B, D]
    const float* rel  = (const float*)d_in[2];  // [NR, 2D]
    const float* Wfcs = (const float*)d_in[3];  // [D, 2D]
    const float* bfcs = (const float*)d_in[4];  // [D]
    float* out = (float*)d_out;                 // [B, D]

    char* ws = (char*)d_ws;
    __bf16* Xb = (__bf16*)ws;                                   // 16 MiB
    __bf16* Rb = (__bf16*)(ws + (size_t)16777216);              //  4 MiB
    __bf16* Wb = (__bf16*)(ws + (size_t)16777216 + 4194304);    //  1 MiB
    float*  R1 = (float*)(ws + (size_t)16777216 + 4194304 + 1048576); // 4 MiB

    // fp32 -> bf16 staging
    pack_stacked_bf16<<<(BATCH * TWOD) / (256 * 4), 256, 0, stream>>>(e1, e2, Xb);
    cvt_bf16x4<<<((NREL * TWOD / 4) + 255) / 256, 256, 0, stream>>>(rel, Rb, NREL * TWOD / 4);
    cvt_bf16x4<<<((DDIM * TWOD / 4) + 255) / 256, 256, 0, stream>>>(Wfcs, Wb, DDIM * TWOD / 4);

    // R1 = relu(R @ W^T + b)  : 4096 tiles, 8 waves/block
    r1_gemm_kernel<<<512, 256, 0, stream>>>(Rb, Wb, bfcs, R1);

    // fused gating: 512 blocks of 16 batch rows each
    gate_fused_kernel<<<BATCH / 16, 256, GATE_LDS_BYTES, stream>>>(Xb, Rb, R1, out);
}
